// NGP_21010980012588
// MI455X (gfx1250) — compile-verified, matrix operand order checked by reading, not run on hardware
//
#include <hip/hip_runtime.h>
#include <math.h>

#define NPTS      524288
#define TBLMASK   0xFFFFFu        // TABLE_SIZE = 1<<20
#define N_LEVELS  12
#define TILES     4               // 16-point tiles per block -> 8192 waves
#define PTS_BLK   (TILES * 16)    // 64 points per block

typedef __attribute__((ext_vector_type(16))) _Float16 v16h;
typedef __attribute__((ext_vector_type(8)))  float    v8f;

struct alignas(8) F2 { float a, b; };

// LEVELS = 4 * 2^(0.43*l), precomputed (matches numpy f64->f32 to ~1e-7 rel)
__constant__ float c_levels[N_LEVELS] = {
  4.00000000f,  5.38893360f,  7.26015330f,  9.78111890f,
  13.17746560f, 17.75312800f, 23.91761680f, 32.22261520f,
  43.41138560f, 58.48526480f, 78.79330960f, 106.15299240f
};

// 16-bit A/B fragment K index for v_wmma_*_16x16x32_f16 (ISA 7.12.2):
// lanes 0-15 hold K = {0..7, 16..23}; lanes 16-31 hold K = {8..15, 24..31}
__device__ __forceinline__ int kmap(int i, int hi) {
  return (i < 8 ? i : i + 8) + (hi << 3);
}

__device__ __forceinline__ float leaky(float v) {
  return fmaxf(v, 0.01f * v);   // slope < 1 => exact leaky_relu
}

__global__ void __launch_bounds__(32)
ngp_fused(const float* __restrict__ x,  const float* __restrict__ grid,
          const float* __restrict__ W1, const float* __restrict__ B1,
          const float* __restrict__ W2, const float* __restrict__ B2,
          const float* __restrict__ W3, const float* __restrict__ B3,
          float* __restrict__ out)
{
  __shared__ float    s_x[48];          // 16 pts * xyz
  __shared__ _Float16 s_feat[16 * 32];  // 16 pts x 24 feats, K-pad to 32,
                                        // K=24 holds constant 1.0 (bias row)
  __shared__ _Float16 s_h1[16 * 64];
  __shared__ _Float16 s_h2[16 * 32];

  const int lane = threadIdx.x;   // 0..31 (wave32)
  const int hi   = lane >> 4;     // half-wave selector
  const int lo   = lane & 15;     // tile row/col within half

  // ---- Resident f16 weight B-fragments (loaded once, reused for all tiles) -
  // Layer-1 bias is folded into the matmul: A carries 1.0 at K=24 (stored in
  // LDS so the A-fragment read stays a contiguous ds_load_b128), and the
  // B-fragment carries b1[n] in row K=24.
  v16h fB1[4];          // W1: 24x64 (+bias row 24), 4 N-tiles, K padded to 32
  v16h fB2[2][2];       // W2: 64x32 -> 2 K-tiles, 2 N-tiles
  v16h fB3;             // W3: 32x2  -> 1 K-tile, N padded 2 -> 16
  #pragma unroll
  for (int t = 0; t < 4; ++t) {
    #pragma unroll
    for (int i = 0; i < 16; ++i) {
      int k = kmap(i, hi);
      float w = (k < 24) ? W1[k * 64 + t * 16 + lo]
              : (k == 24) ? B1[t * 16 + lo] : 0.0f;
      fB1[t][i] = (_Float16)w;
    }
  }
  #pragma unroll
  for (int kt = 0; kt < 2; ++kt)
    #pragma unroll
    for (int nt = 0; nt < 2; ++nt)
      #pragma unroll
      for (int i = 0; i < 16; ++i) {
        int k = kt * 32 + kmap(i, hi);
        fB2[kt][nt][i] = (_Float16)W2[k * 32 + nt * 16 + lo];
      }
  #pragma unroll
  for (int i = 0; i < 16; ++i) {
    int k = kmap(i, hi);
    fB3[i] = (lo < 2) ? (_Float16)W3[k * 2 + lo] : (_Float16)0.0f;
  }
  float bias2[2], bias3;
  #pragma unroll
  for (int t = 0; t < 2; ++t) bias2[t] = B2[t * 16 + lo];
  bias3 = (lo < 2) ? B3[lo] : 0.0f;

  // Init feature pad once (K=24..31): zeros except K=24 = 1.0h (bias row).
  // Done as dword stores: dword 12 of each 16-dword row = {1.0h, 0h}.
  {
    unsigned* sf32 = reinterpret_cast<unsigned*>(s_feat);
    const unsigned v0 = ((lane & 15) == 12) ? 0x00003C00u : 0u;
    #pragma unroll
    for (int i = lane; i < 16 * 16; i += 32) sf32[i] = v0;
  }

  const int base = blockIdx.x * PTS_BLK;

  for (int tile = 0; tile < TILES; ++tile) {
    const int p0 = base + tile * 16;

    // ---- stage coordinates -------------------------------------------------
    #pragma unroll
    for (int i = lane; i < 48; i += 32) s_x[i] = x[p0 * 3 + i];
    __syncthreads();

    // ---- hash-grid encoding: 16 pts x 12 levels = 192 tasks, 6 per lane ----
    // unroll 2 => 16 independent global_load_b64 gathers in flight per lane
    #pragma unroll 2
    for (int t = lane; t < 192; t += 32) {
      const int pt = t & 15, lvl = t >> 4;
      const float lv = c_levels[lvl];
      const float px = s_x[pt * 3 + 0] * lv + (float)lvl;
      const float py = s_x[pt * 3 + 1] * lv + (float)lvl;
      const float pz = s_x[pt * 3 + 2] * lv + (float)lvl;
      const float f0x = floorf(px), f0y = floorf(py), f0z = floorf(pz);
      const int   ix = (int)f0x, iy = (int)f0y, iz = (int)f0z;
      const float fx = px - f0x, fy = py - f0y, fz = pz - f0z;
      const F2* gt = reinterpret_cast<const F2*>(grid + ((size_t)lvl << 21));
      float a0 = 0.0f, a1 = 0.0f;
      #pragma unroll
      for (int c = 0; c < 8; ++c) {
        const int cx = ix + (c & 1), cy = iy + ((c >> 1) & 1), cz = iz + ((c >> 2) & 1);
        const unsigned ux = (unsigned)cx - (cx < 0 ? 1u : 0u);
        const unsigned uy = (unsigned)cy - (cy < 0 ? 1u : 0u);
        const unsigned uz = (unsigned)cz - (cz < 0 ? 1u : 0u);
        const unsigned h = (ux ^ (uy * 2654435761u) ^ (uz * 805459861u)) & TBLMASK;
        const float wx = (c & 1)        ? fx : 1.0f - fx;
        const float wy = ((c >> 1) & 1) ? fy : 1.0f - fy;
        const float wz = ((c >> 2) & 1) ? fz : 1.0f - fz;
        const float w = wx * wy * wz;
        const F2 g = gt[h];               // global_load_b64 gather (L2-resident)
        a0 = fmaf(w, g.a, a0);
        a1 = fmaf(w, g.b, a1);
      }
      s_feat[pt * 32 + lvl * 2 + 0] = (_Float16)a0;
      s_feat[pt * 32 + lvl * 2 + 1] = (_Float16)a1;
    }
    __syncthreads();

    // ---- layer 1: (16x[24+1] pad 32) @ ([24+bias]x64) -> 16x64, 4 WMMAs ----
    v16h A0;
    #pragma unroll
    for (int i = 0; i < 16; ++i) A0[i] = s_feat[lo * 32 + kmap(i, hi)];
    #pragma unroll
    for (int t = 0; t < 4; ++t) {
      v8f acc = {};
      acc = __builtin_amdgcn_wmma_f32_16x16x32_f16(false, A0, false, fB1[t],
                                                   (short)0, acc, false, false);
      #pragma unroll
      for (int r = 0; r < 8; ++r) {
        s_h1[(r + hi * 8) * 64 + t * 16 + lo] = (_Float16)leaky(acc[r]);
      }
    }
    __syncthreads();

    // ---- layer 2: 16x64 @ 64x32 -> 16x32, 2x2 WMMAs ------------------------
    v16h A1a, A1b;
    #pragma unroll
    for (int i = 0; i < 16; ++i) {
      A1a[i] = s_h1[lo * 64 + kmap(i, hi)];
      A1b[i] = s_h1[lo * 64 + 32 + kmap(i, hi)];
    }
    #pragma unroll
    for (int nt = 0; nt < 2; ++nt) {
      v8f acc = {};
      acc = __builtin_amdgcn_wmma_f32_16x16x32_f16(false, A1a, false, fB2[0][nt],
                                                   (short)0, acc, false, false);
      acc = __builtin_amdgcn_wmma_f32_16x16x32_f16(false, A1b, false, fB2[1][nt],
                                                   (short)0, acc, false, false);
      #pragma unroll
      for (int r = 0; r < 8; ++r) {
        s_h2[(r + hi * 8) * 32 + nt * 16 + lo] = (_Float16)leaky(acc[r] + bias2[nt]);
      }
    }
    __syncthreads();

    // ---- layer 3: 16x32 @ 32x2(pad 16) -> 16x2, 1 WMMA ---------------------
    v16h A2;
    #pragma unroll
    for (int i = 0; i < 16; ++i) A2[i] = s_h2[lo * 32 + kmap(i, hi)];
    v8f acc = {};
    acc = __builtin_amdgcn_wmma_f32_16x16x32_f16(false, A2, false, fB3,
                                                 (short)0, acc, false, false);
    if (lo < 2) {
      #pragma unroll
      for (int r = 0; r < 8; ++r) {
        const int p = p0 + r + hi * 8;
        const float v = acc[r] + bias3;
        if (lo == 0) out[p] = v;                              // sigma
        else         out[NPTS + p] = fminf(v, 0.0f) * 0.1f;   // clip(alpha)*0.1
      }
    }
    __syncthreads();
  }
}

extern "C" void kernel_launch(void* const* d_in, const int* in_sizes, int n_in,
                              void* d_out, int out_size, void* d_ws, size_t ws_size,
                              hipStream_t stream) {
  const float* x  = (const float*)d_in[0];
  const float* g  = (const float*)d_in[1];
  const float* W1 = (const float*)d_in[2];
  const float* b1 = (const float*)d_in[3];
  const float* W2 = (const float*)d_in[4];
  const float* b2 = (const float*)d_in[5];
  const float* W3 = (const float*)d_in[6];
  const float* b3 = (const float*)d_in[7];
  float* out = (float*)d_out;

  const int blocks = NPTS / PTS_BLK;  // 8192 single-wave blocks
  ngp_fused<<<blocks, 32, 0, stream>>>(x, g, W1, b1, W2, b2, W3, b3, out);
}
